// AttentionCore64_58085137711581
// MI455X (gfx1250) — matrix operand order repaired and checked by reading, not compiled, not run on hardware
//
#include <hip/hip_runtime.h>
#include <hip/hip_bf16.h>

// Shapes from the reference
#define BATCH 2
#define HEADS 16
#define SEQ   2048
#define DHEAD 64

#define BM    128   // query rows per workgroup
#define BK    64    // key rows per inner tile
#define NWAVE 8     // 256 threads / wave32
#define NTILES (SEQ / BK)
#define LDST  72    // padded LDS row stride (bf16 elems): 36 words -> conflict-free

typedef __attribute__((ext_vector_type(16))) __bf16 bf16x16;
typedef __attribute__((ext_vector_type(8)))  __bf16 bf16x8;
typedef __attribute__((ext_vector_type(4)))  __bf16 bf16x4;
typedef __attribute__((ext_vector_type(2)))  __bf16 bf16x2;
typedef __attribute__((ext_vector_type(8)))  float  f32x8;

__device__ __forceinline__ bf16x16 ld_bf16x16(const __bf16* p) {
    // 16B-aligned LDS loads (two ds_load_b128) + register concat
    bf16x8 lo = *(const bf16x8*)(p);
    bf16x8 hi = *(const bf16x8*)(p + 8);
    return __builtin_shufflevector(lo, hi, 0,1,2,3,4,5,6,7,8,9,10,11,12,13,14,15);
}

__global__ __launch_bounds__(256, 2)
void flash_attn_bf16_wmma(const float* __restrict__ q,
                          const float* __restrict__ k,
                          const float* __restrict__ v,
                          float* __restrict__ out) {
    // K tile bf16 [t][d]; V tile bf16 [d][j] with contraction-permuted j;
    // per-wave P tile [m][j] under the same permutation j = (t&15)*4 + (t>>4).
    __shared__ __attribute__((aligned(128))) __bf16 lds_k[BK * LDST];
    __shared__ __attribute__((aligned(128))) __bf16 lds_v[DHEAD * LDST];
    __shared__ __attribute__((aligned(128))) __bf16 lds_p[NWAVE * 16 * LDST];

    const int tid  = threadIdx.x;
    const int w    = tid >> 5;      // wave id 0..7
    const int lane = tid & 31;
    const int h    = lane >> 4;     // half of wave (ISA fragment split)
    const int n    = lane & 15;     // column / M-row index within fragments

    const int bh = blockIdx.y;                       // fused batch*head
    const int m0 = blockIdx.x * BM + w * 16;         // this wave's 16 query rows

    const float* qb = q + (size_t)bh * SEQ * DHEAD;
    const float* kb = k + (size_t)bh * SEQ * DHEAD;
    const float* vb = v + (size_t)bh * SEQ * DHEAD;

    // ---- Q A-fragments, pre-scaled by log2(e): exp(s) == exp2(s*log2e),
    //      making the softmax exponential a single v_exp_f32. ----
    const float LOG2E = 1.4426950408889634f;
    bf16x16 aq[2];
    {
        const float* qrow = qb + (size_t)(m0 + n) * DHEAD;
#pragma unroll
        for (int kc = 0; kc < 2; ++kc) {
            f32x8 c0 = *(const f32x8*)(qrow + kc * 32 + h * 16);
            f32x8 c1 = *(const f32x8*)(qrow + kc * 32 + h * 16 + 8);
#pragma unroll
            for (int e = 0; e < 8; ++e) {
                aq[kc][e]     = (__bf16)(c0[e] * LOG2E);
                aq[kc][e + 8] = (__bf16)(c1[e] * LOG2E);
            }
        }
    }

    // ---- Unnormalized accumulators (reference uses raw exp; |s| stays
    //      small enough that exp and row sums fit comfortably in f32). ----
    f32x8 o[4];
#pragma unroll
    for (int dt = 0; dt < 4; ++dt)
#pragma unroll
        for (int r = 0; r < 8; ++r) o[dt][r] = 0.0f;

    float lsum[8];
#pragma unroll
    for (int r = 0; r < 8; ++r) lsum[r] = 0.0f;

    __bf16* pw = lds_p + w * (16 * LDST);

    // Staging coordinates (fixed per thread)
    const int kt_t  = tid >> 2;             // 0..63
    const int kt_d0 = (tid & 3) * 16;       // 0,16,32,48
    const int vt_g  = (tid >> 3) >> 4;      // 0..1
    const int vt_x  = (tid >> 3) & 15;      // 0..15
    const int vt_d0 = (tid & 7) * 8;        // 0..56
    // physical V rows handled by this thread: ra = 32g + x, rb = ra + 16
    const int vt_ra = vt_g * 32 + vt_x;

    // ---- Software pipeline: pointers to the next tile to fetch ----
    const float* kp  = kb + (size_t)kt_t * DHEAD + kt_d0;
    const float* vpa = vb + (size_t)vt_ra * DHEAD + vt_d0;

    f32x8 kx0, kx1, vxa, vxb;
    kx0 = *(const f32x8*)(kp);
    kx1 = *(const f32x8*)(kp + 8);
    vxa = *(const f32x8*)(vpa);
    vxb = *(const f32x8*)(vpa + 16 * DHEAD);
    kp  += BK * DHEAD;
    vpa += BK * DHEAD;

    for (int it = 0; it < NTILES; ++it) {
        // ---- Drain staged registers into LDS (one f32->bf16 convert) ----
        {
            bf16x8 y0, y1;
#pragma unroll
            for (int e = 0; e < 8; ++e) { y0[e] = (__bf16)kx0[e]; y1[e] = (__bf16)kx1[e]; }
            *(bf16x8*)&lds_k[kt_t * LDST + kt_d0]     = y0;
            *(bf16x8*)&lds_k[kt_t * LDST + kt_d0 + 8] = y1;
            // permuted V: rows (32g+x, 32g+16+x) -> positions (4x+2g, 4x+2g+1)
#pragma unroll
            for (int j = 0; j < 8; ++j) {
                bf16x2 pkt; pkt[0] = (__bf16)vxa[j]; pkt[1] = (__bf16)vxb[j];
                *(bf16x2*)&lds_v[(vt_d0 + j) * LDST + vt_x * 4 + 2 * vt_g] = pkt;
            }
        }
        __syncthreads();

        // ---- Prefetch next tile's K/V (overlaps with the WMMA work) ----
        kx0 = *(const f32x8*)(kp);
        kx1 = *(const f32x8*)(kp + 8);
        vxa = *(const f32x8*)(vpa);
        vxb = *(const f32x8*)(vpa + 16 * DHEAD);
        {
            const int adv = (it + 2 < NTILES) ? (BK * DHEAD) : 0;  // uniform
            kp  += adv;
            vpa += adv;
        }

        // ---- S = (Q*log2e) * K^T : kc-outer so 4 consecutive WMMAs share A ----
        f32x8 s[4];
#pragma unroll
        for (int nt = 0; nt < 4; ++nt)
#pragma unroll
            for (int r = 0; r < 8; ++r) s[nt][r] = 0.0f;
#pragma unroll
        for (int kc = 0; kc < 2; ++kc) {
#pragma unroll
            for (int nt = 0; nt < 4; ++nt) {
                bf16x16 bk = ld_bf16x16(&lds_k[(nt * 16 + n) * LDST + kc * 32 + h * 16]);
                s[nt] = __builtin_amdgcn_wmma_f32_16x16x32_bf16(
                            false, aq[kc], false, bk, (short)0, s[nt], false, false);
            }
        }

        // ---- P = exp2(S); pack 4 values -> one b64 LDS store per row ----
#pragma unroll
        for (int r = 0; r < 8; ++r) {
            float p0 = __builtin_amdgcn_exp2f(s[0][r]);
            float p1 = __builtin_amdgcn_exp2f(s[1][r]);
            float p2 = __builtin_amdgcn_exp2f(s[2][r]);
            float p3 = __builtin_amdgcn_exp2f(s[3][r]);
            lsum[r] += (p0 + p1) + (p2 + p3);
            bf16x4 pk;
            pk[0] = (__bf16)p0; pk[1] = (__bf16)p1;
            pk[2] = (__bf16)p2; pk[3] = (__bf16)p3;
            // permuted column position: cols (n, n+16, n+32, n+48) -> 4n..4n+3
            *(bf16x4*)&pw[(r + 8 * h) * LDST + n * 4] = pk;
        }

        // ---- Re-read P as A-fragments (linear in permuted positions) ----
        bf16x16 pa[2];
#pragma unroll
        for (int tc = 0; tc < 2; ++tc) {
            bf16x8 c0 = *(const bf16x8*)&pw[n * LDST + tc * 32 + h * 16];
            bf16x8 c1 = *(const bf16x8*)&pw[n * LDST + tc * 32 + h * 16 + 8];
            pa[tc] = __builtin_shufflevector(c0, c1,
                        0,1,2,3,4,5,6,7,8,9,10,11,12,13,14,15);
        }

        // ---- O += P * V : tc-outer so 4 consecutive WMMAs share A ----
#pragma unroll
        for (int tc = 0; tc < 2; ++tc) {
#pragma unroll
            for (int dt = 0; dt < 4; ++dt) {
                bf16x16 bv = ld_bf16x16(&lds_v[(dt * 16 + n) * LDST + tc * 32 + h * 16]);
                o[dt] = __builtin_amdgcn_wmma_f32_16x16x32_bf16(
                            false, pa[tc], false, bv, (short)0, o[dt], false, false);
            }
        }
        __syncthreads();   // protect lds_k / lds_v before next staging pass
    }

    // ---- Epilogue: one cross-lane row-sum reduction, divide, store ----
    float* ob = out + ((size_t)bh * SEQ + m0) * DHEAD;
#pragma unroll
    for (int r = 0; r < 8; ++r) {
        float l = lsum[r];
#pragma unroll
        for (int mk = 1; mk < 16; mk <<= 1)
            l += __shfl_xor(l, mk, 32);          // reduce over the 16 lanes of this half
        const float inv = 1.0f / l;
#pragma unroll
        for (int dt = 0; dt < 4; ++dt)
            ob[(size_t)(r + 8 * h) * DHEAD + dt * 16 + n] = o[dt][r] * inv;
    }
}

extern "C" void kernel_launch(void* const* d_in, const int* in_sizes, int n_in,
                              void* d_out, int out_size, void* d_ws, size_t ws_size,
                              hipStream_t stream) {
    const float* q = (const float*)d_in[0];
    const float* k = (const float*)d_in[1];
    const float* v = (const float*)d_in[2];
    float* out = (float*)d_out;

    dim3 grid(SEQ / BM, BATCH * HEADS, 1);   // (16, 32)
    dim3 block(32 * NWAVE, 1, 1);            // 256 threads = 8 wave32
    flash_attn_bf16_wmma<<<grid, block, 0, stream>>>(q, k, v, out);
}
